// SymAsymRNN_10728828305912
// MI455X (gfx1250) — matrix operand order repaired and checked by reading, not compiled, not run on hardware
//
#include <hip/hip_runtime.h>
#include <hip/hip_bf16.h>
#include <math.h>

#define DEV __device__ __forceinline__

typedef __attribute__((ext_vector_type(16))) __bf16 bf16x16;
typedef __attribute__((ext_vector_type(8)))  float  f32x8;

constexpr int B_ = 32, T_ = 2048, D_ = 256, H_ = 512, O_ = 256;
constexpr int BT_ = B_ * T_;

union B16Vec { bf16x16 v; unsigned int u[8]; int4 q[2]; };
union F8     { f32x8 v; float f[8]; };

// float -> bf16 (round to nearest even), bit-level so we never do __bf16 math
DEV unsigned int f2bf(float f) {
  unsigned int u = __float_as_uint(f);
  return (u + 0x7FFFu + ((u >> 16) & 1u)) >> 16;
}
DEV unsigned int   pack2bf(float lo, float hi) { return f2bf(lo) | (f2bf(hi) << 16); }
DEV unsigned short f2bfs(float f)              { return (unsigned short)f2bf(f); }

DEV f32x8 wmma_bf16(bf16x16 a, bf16x16 b, f32x8 c) {
  // D = A(16x32 bf16) * B(32x16 bf16) + C(16x16 f32)
  return __builtin_amdgcn_wmma_f32_16x16x32_bf16(false, a, false, b, (short)0, c,
                                                 false, false);
}

// A-operand: 16x32 bf16 tile, row-major in LDS.
// A k order used here (contiguous, B packing matches): lanes 0-15 hold row M=lane, K=0..15;
// lanes 16-31 hold row M=lane-16, K=16..31. Two ds_load_b128 per lane.
DEV bf16x16 load_a_lds(const unsigned short* tile, int strideElems, int lane) {
  int row  = lane & 15;
  int koff = (lane & 16) ? 16 : 0;
  const unsigned short* p = tile + row * strideElems + koff;
  B16Vec r;
  r.q[0] = *reinterpret_cast<const int4*>(p);
  r.q[1] = *reinterpret_cast<const int4*>(p + 8);
  return r.v;
}

// B-operand from pre-swizzled packed weights: 32 contiguous bytes per lane.
DEV bf16x16 load_b_packed(const unsigned int* pack, int tileIdx, int lane) {
  const unsigned int* p = pack + (size_t)(tileIdx * 32 + lane) * 8;
  B16Vec r;
  r.q[0] = *reinterpret_cast<const int4*>(p);
  r.q[1] = *reinterpret_cast<const int4*>(p + 4);
  return r.v;
}

// ---------------------------------------------------------------------------
// Pack W = lam*sym(S) + (1-lam)*asym(A), transposed (B[k,n] = W[n,k]),
// directly into the WMMA B-register layout. One dword (bf16 pair) per thread.
// Tile index = kt*32 + nt ; K=H=512 (16 k-tiles), N=H=512 (32 n-tiles).
__global__ __launch_bounds__(256) void pack_w_kernel(
    const float* __restrict__ S, const float* __restrict__ A,
    const float* __restrict__ lam_raw, unsigned int* __restrict__ wtPack) {
  int tid  = blockIdx.x * blockDim.x + threadIdx.x;
  int v    = tid & 7;
  int lane = (tid >> 3) & 31;
  int tile = tid >> 8;                 // 0..511
  int nt = tile & 31, kt = tile >> 5;
  int n = nt * 16 + (lane & 15);
  int k = kt * 32 + ((lane & 16) ? 16 : 0) + v * 2;
  float lam = 1.f / (1.f + __expf(-lam_raw[0]));
  float w0 = lam * 0.5f * (S[n * H_ + k]       + S[k * H_ + n]) +
             (1.f - lam) * 0.5f * (A[n * H_ + k]       - A[k * H_ + n]);
  float w1 = lam * 0.5f * (S[n * H_ + k + 1]   + S[(k + 1) * H_ + n]) +
             (1.f - lam) * 0.5f * (A[n * H_ + k + 1]   - A[(k + 1) * H_ + n]);
  wtPack[(size_t)(tile * 32 + lane) * 8 + v] = pack2bf(w0, w1);
}

// Pack a row-major [N,K] fp32 weight as its transpose B[k,n]=src[n*K+k] into
// the WMMA B-register layout (used for W_in [H,D] and W_out [O,H]).
__global__ __launch_bounds__(256) void pack_b_kernel(
    const float* __restrict__ src, unsigned int* __restrict__ dst, int K, int NT) {
  int tid  = blockIdx.x * blockDim.x + threadIdx.x;
  int v    = tid & 7;
  int lane = (tid >> 3) & 31;
  int tile = tid >> 8;
  int nt = tile % NT, kt = tile / NT;
  int n = nt * 16 + (lane & 15);
  int k = kt * 32 + ((lane & 16) ? 16 : 0) + v * 2;
  dst[(size_t)(tile * 32 + lane) * 8 + v] =
      pack2bf(src[(size_t)n * K + k], src[(size_t)n * K + k + 1]);
}

// ---------------------------------------------------------------------------
// xp = x @ W_in^T + b_in, written fp32 in [T,B,H] layout for the scan.
// One block = 16 rows of BT, all 32 N-tiles (4 per wave). K=256 (8 k-tiles).
__global__ __launch_bounds__(256) void xp_gemm_kernel(
    const float* __restrict__ x, const unsigned int* __restrict__ winPack,
    const float* __restrict__ b_in, float* __restrict__ xp) {
  __shared__ __align__(16) unsigned short a_bf[16 * 264];  // padded stride
  int tid = threadIdx.x;
  int m0  = blockIdx.x * 16;
  {  // stage 16x256 of x as bf16
    int row = tid >> 4, c0 = (tid & 15) * 16;
    const float* src = x + (size_t)(m0 + row) * D_ + c0;
    unsigned short* dstp = a_bf + row * 264 + c0;
#pragma unroll
    for (int j = 0; j < 16; ++j) dstp[j] = f2bfs(src[j]);
  }
  __syncthreads();
  int w = tid >> 5, lane = tid & 31;
  F8 acc[4];
#pragma unroll
  for (int i = 0; i < 4; ++i) { f32x8 z = {}; acc[i].v = z; }
#pragma unroll
  for (int kt = 0; kt < 8; ++kt) {
    bf16x16 a = load_a_lds(a_bf + kt * 32, 264, lane);
#pragma unroll
    for (int i = 0; i < 4; ++i) {
      int nt = w * 4 + i;
      acc[i].v = wmma_bf16(a, load_b_packed(winPack, kt * 32 + nt, lane), acc[i].v);
    }
  }
  int rbase = (lane & 16) ? 8 : 0;
#pragma unroll
  for (int i = 0; i < 4; ++i) {
    int n = (w * 4 + i) * 16 + (lane & 15);
    float bn = b_in[n];
#pragma unroll
    for (int r = 0; r < 8; ++r) {
      int m = m0 + rbase + r;          // flat bt index
      int b = m >> 11, t = m & (T_ - 1);
      xp[((size_t)t * B_ + b) * H_ + n] = acc[i].f[r] + bn;
    }
  }
}

// ---------------------------------------------------------------------------
// Sequential scan: h = tanh(xp_t + h @ W^T), one persistent 32-wave workgroup.
// Wave w owns N-tile w; its W slice (16 k-tiles, 512B/lane) stays in VGPRs.
// h lives in LDS as bf16 with padded row stride 520 to spread banks.
__global__ __launch_bounds__(1024) void scan_kernel(
    const float* __restrict__ h0, const unsigned int* __restrict__ wtPack,
    const float* __restrict__ xp, float* __restrict__ hs) {
  __shared__ __align__(16) unsigned short h_bf[32 * 520];
  int tid = threadIdx.x, w = tid >> 5, lane = tid & 31;

  bf16x16 wb[16];  // register-resident weight slice for this wave's N-tile
#pragma unroll
  for (int kt = 0; kt < 16; ++kt)
    wb[kt] = load_b_packed(wtPack, kt * 32 + w, lane);

  {  // h0 -> LDS (bf16)
    int b = tid >> 5, c0 = (tid & 31) * 16;
    const float* src = h0 + (size_t)b * H_ + c0;
    unsigned short* dstp = h_bf + b * 520 + c0;
#pragma unroll
    for (int j = 0; j < 16; ++j) dstp[j] = f2bfs(src[j]);
  }
  __syncthreads();

  int n     = w * 16 + (lane & 15);
  int rbase = (lane & 16) ? 8 : 0;

  for (int t = 0; t < T_; ++t) {
    const float* xpt = xp + (size_t)t * B_ * H_;
    F8 c0, c1;
#pragma unroll
    for (int r = 0; r < 8; ++r) {            // accumulators start at xp_t
      c0.f[r] = xpt[(rbase + r) * H_ + n];
      c1.f[r] = xpt[(rbase + r + 16) * H_ + n];
    }
    if (t + 1 < T_)                           // global_prefetch_b8 next step
      __builtin_prefetch(xpt + (size_t)B_ * H_ + rbase * H_ + n, 0, 0);

#pragma unroll
    for (int kt = 0; kt < 16; ++kt) {
      bf16x16 a0 = load_a_lds(h_bf + kt * 32, 520, lane);            // rows 0-15
      bf16x16 a1 = load_a_lds(h_bf + 16 * 520 + kt * 32, 520, lane); // rows 16-31
      c0.v = wmma_bf16(a0, wb[kt], c0.v);
      c1.v = wmma_bf16(a1, wb[kt], c1.v);
    }

    float t0[8], t1[8];
#pragma unroll
    for (int r = 0; r < 8; ++r) {
      t0[r] = tanhf(c0.f[r]);
      t1[r] = tanhf(c1.f[r]);
      int b0 = rbase + r, b1 = b0 + 16;
      hs[((size_t)b0 * T_ + t) * H_ + n] = t0[r];   // [B,T,H] output
      hs[((size_t)b1 * T_ + t) * H_ + n] = t1[r];
    }
    __syncthreads();  // everyone done reading old h
#pragma unroll
    for (int r = 0; r < 8; ++r) {
      h_bf[(rbase + r) * 520 + n]      = f2bfs(t0[r]);
      h_bf[(rbase + r + 16) * 520 + n] = f2bfs(t1[r]);
    }
    __syncthreads();  // new h visible
  }
}

// ---------------------------------------------------------------------------
// ys = softmax(hs @ W_out^T + b_out), fully parallel. One block = 16 rows,
// all 16 N-tiles of O=256 (2 per wave), then block-local row softmax.
__global__ __launch_bounds__(256) void out_gemm_kernel(
    const float* __restrict__ hs, const unsigned int* __restrict__ woutPack,
    const float* __restrict__ b_out, float* __restrict__ ys) {
  __shared__ __align__(16) unsigned short a_bf[16 * 520];
  __shared__ float lg[16 * 272];
  __shared__ float red[16 * 16];
  __shared__ float rowM[16], rowS[16];
  int tid = threadIdx.x, w = tid >> 5, lane = tid & 31;
  int m0  = blockIdx.x * 16;
  {  // stage 16x512 of hs as bf16
    int row = tid >> 4, c0 = (tid & 15) * 32;
    const float* src = hs + (size_t)(m0 + row) * H_ + c0;
    unsigned short* dstp = a_bf + row * 520 + c0;
#pragma unroll
    for (int j = 0; j < 32; ++j) dstp[j] = f2bfs(src[j]);
  }
  __syncthreads();

  F8 c0v, c1v;
  { f32x8 z = {}; c0v.v = z; c1v.v = z; }
  int nt0 = w * 2, nt1 = w * 2 + 1;
#pragma unroll
  for (int kt = 0; kt < 16; ++kt) {
    bf16x16 a = load_a_lds(a_bf + kt * 32, 520, lane);
    c0v.v = wmma_bf16(a, load_b_packed(woutPack, kt * 16 + nt0, lane), c0v.v);
    c1v.v = wmma_bf16(a, load_b_packed(woutPack, kt * 16 + nt1, lane), c1v.v);
  }
  int rbase = (lane & 16) ? 8 : 0;
  {
    int n0 = nt0 * 16 + (lane & 15), n1 = nt1 * 16 + (lane & 15);
    float bo0 = b_out[n0], bo1 = b_out[n1];
#pragma unroll
    for (int r = 0; r < 8; ++r) {
      lg[(rbase + r) * 272 + n0] = c0v.f[r] + bo0;
      lg[(rbase + r) * 272 + n1] = c1v.f[r] + bo1;
    }
  }
  __syncthreads();

  // softmax over 256 cols/row: 16 threads per row, 16 cols each
  int row = tid >> 4, seg = tid & 15;
  const float* lr = lg + row * 272 + seg * 16;
  float lmax = lr[0];
#pragma unroll
  for (int j = 1; j < 16; ++j) lmax = fmaxf(lmax, lr[j]);
  red[row * 16 + seg] = lmax;
  __syncthreads();
  if (seg == 0) {
    float m = red[row * 16];
#pragma unroll
    for (int j = 1; j < 16; ++j) m = fmaxf(m, red[row * 16 + j]);
    rowM[row] = m;
  }
  __syncthreads();
  float m = rowM[row];
  float lsum = 0.f;
#pragma unroll
  for (int j = 0; j < 16; ++j) lsum += __expf(lr[j] - m);
  red[row * 16 + seg] = lsum;
  __syncthreads();
  if (seg == 0) {
    float s = 0.f;
#pragma unroll
    for (int j = 0; j < 16; ++j) s += red[row * 16 + j];
    rowS[row] = s;
  }
  __syncthreads();
  float inv = 1.f / rowS[row];
  float* yrow = ys + (size_t)(m0 + row) * O_ + seg * 16;
#pragma unroll
  for (int j = 0; j < 16; ++j) yrow[j] = __expf(lr[j] - m) * inv;
}

// ---------------------------------------------------------------------------
extern "C" void kernel_launch(void* const* d_in, const int* in_sizes, int n_in,
                              void* d_out, int out_size, void* d_ws, size_t ws_size,
                              hipStream_t stream) {
  (void)in_sizes; (void)n_in; (void)out_size; (void)ws_size;
  const float* x     = (const float*)d_in[0];
  const float* h0    = (const float*)d_in[1];
  const float* S     = (const float*)d_in[2];
  const float* A     = (const float*)d_in[3];
  const float* lam   = (const float*)d_in[4];
  const float* W_in  = (const float*)d_in[5];
  const float* b_in  = (const float*)d_in[6];
  const float* W_out = (const float*)d_in[7];
  const float* b_out = (const float*)d_in[8];

  // Workspace: packed weights (1 MB total) + xp[T,B,H] fp32 (128 MB)
  uint8_t* wsb = (uint8_t*)d_ws;
  unsigned int* wtPack   = (unsigned int*)(wsb);
  unsigned int* winPack  = (unsigned int*)(wsb + (512u << 10));
  unsigned int* woutPack = (unsigned int*)(wsb + (768u << 10));
  float*        xp       = (float*)(wsb + (1u << 20));

  float* ys = (float*)d_out;                 // [B,T,O]
  float* hs = ys + (size_t)BT_ * O_;         // [B,T,H]

  pack_w_kernel<<<512, 256, 0, stream>>>(S, A, lam, wtPack);
  pack_b_kernel<<<256, 256, 0, stream>>>(W_in,  winPack,  D_, 32);
  pack_b_kernel<<<256, 256, 0, stream>>>(W_out, woutPack, H_, 16);
  xp_gemm_kernel<<<BT_ / 16, 256, 0, stream>>>(x, winPack, b_in, xp);
  scan_kernel<<<1, 1024, 0, stream>>>(h0, wtPack, xp, hs);
  out_gemm_kernel<<<BT_ / 16, 256, 0, stream>>>(hs, woutPack, b_out, ys);
}
